// Attention_15994458211117
// MI455X (gfx1250) — matrix operand order repaired and checked by reading, not compiled, not run on hardware
//
#include <hip/hip_runtime.h>
#include <hip/hip_bf16.h>
#include <math.h>

// Problem sizes (fixed by the reference)
#define B_   64
#define T_   2048
#define ENC_ 512
#define DEC_ 512
#define U_   128

typedef __attribute__((ext_vector_type(16))) __bf16 v16bf;
typedef __attribute__((ext_vector_type(8)))  float  v8f;

union Frag16 {
    uint4 u[2];   // 32 bytes
    v16bf v;      // 16 x bf16
};

__device__ __forceinline__ unsigned short f2bf(float f) {
    unsigned int u = __float_as_uint(f);
    unsigned int r = u + 0x7FFFu + ((u >> 16) & 1u);   // round-to-nearest-even
    return (unsigned short)(r >> 16);
}

// ---------------------------------------------------------------------------
// Workspace layout (bytes)
//   qq      : B*U  f32          @ 0        (32 KB)   q = dh@Wa + ba + bb
//   WbT     : U*ENC bf16        @ 32768    (128 KB)  Wb transposed, bf16
//   score   : B*T  f32          @ 163840   (512 KB)
//   partial : B*8*ENC f32       @ 688128   (1 MB)
// ---------------------------------------------------------------------------
#define OFF_QQ      0
#define OFF_WBT     32768
#define OFF_SCORE   163840
#define OFF_PARTIAL 688128

// ===========================================================================
// Kernel 1: qq[b][u] = dot(dh[b,:], Wa[:,u]) + ba[u] + bb[u]
// ===========================================================================
__global__ void prep_q_kernel(const float* __restrict__ dh,
                              const float* __restrict__ Wa,
                              const float* __restrict__ ba,
                              const float* __restrict__ bb,
                              float* __restrict__ qq) {
    __shared__ float dh_s[DEC_];
    const int b = blockIdx.x;
    const int u = threadIdx.x;
    #pragma unroll
    for (int i = 0; i < DEC_ / U_; ++i)
        dh_s[u + i * U_] = dh[b * DEC_ + u + i * U_];
    __syncthreads();
    float acc = 0.f;
    for (int k = 0; k < DEC_; ++k)
        acc = fmaf(dh_s[k], Wa[k * U_ + u], acc);
    qq[b * U_ + u] = acc + ba[u] + bb[u];
}

// ===========================================================================
// Kernel 2: WbT[n][k] = bf16(Wb[k][n])
// ===========================================================================
__global__ void prep_wbt_kernel(const float* __restrict__ Wb,
                                unsigned short* __restrict__ wbt) {
    const int idx = blockIdx.x * 256 + threadIdx.x;   // idx = k*U_ + n
    const int k = idx >> 7;
    const int n = idx & (U_ - 1);
    wbt[n * ENC_ + k] = f2bf(Wb[idx]);
}

// ===========================================================================
// Kernel 3: WMMA score kernel.
//   block = 512 (16 waves), grid = (T_/128, B_)
//   Wave w: M-tile = w&7 (16 rows of t), N-half = w>>3 (4 of 8 u-tiles)
//   k_proj = enc_tile(128x512 bf16) @ WbT^T, f32 accum
//   score[t] = sum_u tanh(qq[u] + k_proj[t,u]) * Wv[u] + bv
// LDS (dynamic):
//   ldsB : 128 x 520 bf16  (full Wb^T, async-staged once)  133120 B
//   bufA : 2 x (128 x 136) bf16  (double-buffered chunk)    69632 B
//   redS : 2 x 128 f32  (cross-wave u-half combine)          1024 B
// ===========================================================================
#define TT    128        // T rows per block
#define KC    128        // K chunk staged per iteration
#define LDA   136        // A row stride (bf16), 272 B, 16B aligned
#define LDBF  520        // B row stride (bf16), 1040 B, 16B aligned

#define LDSB_US   (U_ * LDBF)            // 66560 ushorts
#define BUFA_US   (TT * LDA)             // 17408 ushorts
#define REDS_BYTE ((LDSB_US + 2 * BUFA_US) * 2)
#define LDS_BYTES (REDS_BYTE + 2 * TT * 4)

__global__ void __launch_bounds__(512)
score_wmma_kernel(const float* __restrict__ enc,
                  const float* __restrict__ qq,
                  const float* __restrict__ Wv,
                  const float* __restrict__ bv,
                  const unsigned short* __restrict__ wbt,
                  float* __restrict__ score) {
    extern __shared__ unsigned short smem[];
    unsigned short* ldsB  = smem;
    unsigned short* bufA0 = smem + LDSB_US;
    unsigned short* bufA1 = bufA0 + BUFA_US;
    float*          redS  = (float*)((char*)smem + REDS_BYTE);

    const int tid  = threadIdx.x;
    const int wave = tid >> 5;
    const int lane = tid & 31;
    const int mw   = wave & 7;        // M-tile
    const int nw   = wave >> 3;       // N half (0/1)
    const int b    = blockIdx.y;
    const int t0   = blockIdx.x * TT;

    const size_t encBase = ((size_t)b * T_ + t0) * ENC_;

    // Per-thread staging offsets (8 float4 per thread per chunk)
    int gOff[8], lOff[8];
    #pragma unroll
    for (int i = 0; i < 8; ++i) {
        const int idx4 = tid + i * 512;        // 4096 float4 per chunk
        const int row  = idx4 >> 5;            // 32 float4 per 128-wide row
        const int c4   = idx4 & 31;
        gOff[i] = row * ENC_ + c4 * 4;         // + kc*KC later
        lOff[i] = row * LDA + c4 * 4;
    }

    // ---- issue global loads for A chunk 0 ----
    float4 regs[8];
    #pragma unroll
    for (int i = 0; i < 8; ++i)
        regs[i] = *(const float4*)(enc + encBase + gOff[i]);

    // ---- stage full Wb^T via async global->LDS copies (no VGPR round trip)
    //      128 KB = 8192 x 16B chunks; one b128 copy per lane per issue.
    {
        const unsigned ldsBBase = (unsigned)(uintptr_t)(void*)ldsB;
        #pragma unroll
        for (int i = 0; i < 16; ++i) {
            const int idx = tid + i * 512;       // 16B-chunk index, 8192 total
            const int n   = idx >> 6;            // 64 chunks per 512-elem row
            const int c   = idx & 63;            // chunk within row (8 ushorts)
            const unsigned lds_addr = ldsBBase + (unsigned)(n * LDBF + c * 8) * 2u;
            const unsigned long long gaddr =
                (unsigned long long)(uintptr_t)(wbt + n * ENC_ + c * 8);
            asm volatile("global_load_async_to_lds_b128 %0, %1, off"
                         :: "v"(lds_addr), "v"(gaddr) : "memory");
        }
    }

    // ---- convert + store A chunk 0 (waits LOADcnt only) ----
    #pragma unroll
    for (int i = 0; i < 8; ++i) {
        ushort4 h;
        h.x = f2bf(regs[i].x); h.y = f2bf(regs[i].y);
        h.z = f2bf(regs[i].z); h.w = f2bf(regs[i].w);
        *(ushort4*)(bufA0 + lOff[i]) = h;
    }
    // all async B copies of this wave must be in LDS before anyone reads ldsB
    asm volatile("s_wait_asynccnt 0x0" ::: "memory");
    __syncthreads();

    v8f acc[4];
    const v8f vzero = {0.f, 0.f, 0.f, 0.f, 0.f, 0.f, 0.f, 0.f};
    #pragma unroll
    for (int j = 0; j < 4; ++j) acc[j] = vzero;

    const int aRowOff = (mw * 16 + (lane & 15)) * LDA + ((lane & 16) ? 8 : 0);
    const int cN      = lane & 15;
    const int bKH     = (lane & 16) ? 8 : 0;

    #pragma unroll 1
    for (int kc = 0; kc < ENC_ / KC; ++kc) {
        // issue global loads for next chunk (overlap with WMMA below)
        if (kc < 3) {
            #pragma unroll
            for (int i = 0; i < 8; ++i)
                regs[i] = *(const float4*)(enc + encBase + (kc + 1) * KC + gOff[i]);
        }

        const unsigned short* bufA = (kc & 1) ? bufA1 : bufA0;
        #pragma unroll
        for (int ks = 0; ks < 4; ++ks) {
            Frag16 fa;
            fa.u[0] = *(const uint4*)(bufA + aRowOff + ks * 32);
            fa.u[1] = *(const uint4*)(bufA + aRowOff + ks * 32 + 16);
            Frag16 fb[4];
            #pragma unroll
            for (int j = 0; j < 4; ++j) {
                const int bOff = ((nw * 4 + j) * 16 + cN) * LDBF
                               + kc * KC + ks * 32 + bKH;
                fb[j].u[0] = *(const uint4*)(ldsB + bOff);
                fb[j].u[1] = *(const uint4*)(ldsB + bOff + 16);
            }
            #pragma unroll
            for (int j = 0; j < 4; ++j)
                acc[j] = __builtin_amdgcn_wmma_f32_16x16x32_bf16(
                    false, fa.v, false, fb[j].v, (short)0, acc[j], false, false);
        }

        // convert + store next chunk into the other buffer
        if (kc < 3) {
            unsigned short* bufN = (kc & 1) ? bufA0 : bufA1;
            #pragma unroll
            for (int i = 0; i < 8; ++i) {
                ushort4 h;
                h.x = f2bf(regs[i].x); h.y = f2bf(regs[i].y);
                h.z = f2bf(regs[i].z); h.w = f2bf(regs[i].w);
                *(ushort4*)(bufN + lOff[i]) = h;
            }
        }
        __syncthreads();
    }

    // ---- epilogue: tanh(qq + k) * Wv, reduce over this wave's u-half ----
    float qv[4], wv[4];
    #pragma unroll
    for (int j = 0; j < 4; ++j) {
        const int u = (nw * 4 + j) * 16 + cN;
        qv[j] = qq[b * U_ + u];
        wv[j] = Wv[u];
    }
    float sAcc[8];
    #pragma unroll
    for (int r = 0; r < 8; ++r) sAcc[r] = 0.f;
    #pragma unroll
    for (int j = 0; j < 4; ++j) {
        #pragma unroll
        for (int r = 0; r < 8; ++r)
            sAcc[r] = fmaf(tanhf(qv[j] + acc[j][r]), wv[j], sAcc[r]);
    }
    // butterfly over the 16-lane column group (C layout: N = lane&15)
    #pragma unroll
    for (int r = 0; r < 8; ++r) {
        #pragma unroll
        for (int m = 1; m < 16; m <<= 1)
            sAcc[r] += __shfl_xor(sAcc[r], m, 32);
    }
    if ((lane & 15) == 0) {
        const int rbase = mw * 16 + ((lane & 16) ? 8 : 0);
        #pragma unroll
        for (int r = 0; r < 8; ++r)
            redS[nw * TT + rbase + r] = sAcc[r];
    }
    __syncthreads();
    // combine the two u-halves, add bv, write scores
    if (tid < TT) {
        const float s = redS[tid] + redS[TT + tid] + bv[0];
        score[b * T_ + t0 + tid] = s;
    }
}

// ===========================================================================
// Kernel 4: per-batch softmax over T; writes attn to its d_out slice.
// ===========================================================================
__global__ void softmax_kernel(const float* __restrict__ score,
                               float* __restrict__ attn) {
    __shared__ float red[8];
    const int b = blockIdx.x;
    const int tid = threadIdx.x;
    const int wave = tid >> 5;
    const int lane = tid & 31;

    float s[8];
    #pragma unroll
    for (int i = 0; i < 8; ++i) s[i] = score[b * T_ + tid + i * 256];

    float m = s[0];
    #pragma unroll
    for (int i = 1; i < 8; ++i) m = fmaxf(m, s[i]);
    #pragma unroll
    for (int o = 1; o < 32; o <<= 1) m = fmaxf(m, __shfl_xor(m, o, 32));
    if (lane == 0) red[wave] = m;
    __syncthreads();
    float gm = red[0];
    #pragma unroll
    for (int w = 1; w < 8; ++w) gm = fmaxf(gm, red[w]);
    __syncthreads();

    float e[8], sum = 0.f;
    #pragma unroll
    for (int i = 0; i < 8; ++i) { e[i] = __expf(s[i] - gm); sum += e[i]; }
    #pragma unroll
    for (int o = 1; o < 32; o <<= 1) sum += __shfl_xor(sum, o, 32);
    if (lane == 0) red[wave] = sum;
    __syncthreads();
    float gs = 0.f;
    #pragma unroll
    for (int w = 0; w < 8; ++w) gs += red[w];
    const float inv = 1.0f / gs;

    #pragma unroll
    for (int i = 0; i < 8; ++i)
        attn[b * T_ + tid + i * 256] = e[i] * inv;
}

// ===========================================================================
// Kernel 5: context partials: partial[b][slice][e] = sum_{t in slice} a*enc
// ===========================================================================
__global__ void ctx_partial_kernel(const float* __restrict__ enc,
                                   const float* __restrict__ attn,
                                   float* __restrict__ partial) {
    const int slice = blockIdx.x;
    const int b     = blockIdx.y;
    const int tid   = threadIdx.x;
    float ax = 0.f, ay = 0.f;
    const int tBeg = slice * (T_ / 8);
    for (int t = tBeg; t < tBeg + (T_ / 8); ++t) {
        const float w  = attn[b * T_ + t];
        const float2 e = *(const float2*)(enc + ((size_t)b * T_ + t) * ENC_ + tid * 2);
        ax = fmaf(w, e.x, ax);
        ay = fmaf(w, e.y, ay);
    }
    float2 o; o.x = ax; o.y = ay;
    *(float2*)(partial + ((size_t)(b * 8 + slice)) * ENC_ + tid * 2) = o;
}

// ===========================================================================
// Kernel 6: reduce 8 partials -> context.
// ===========================================================================
__global__ void ctx_reduce_kernel(const float* __restrict__ partial,
                                  float* __restrict__ ctx) {
    const int b = blockIdx.x;
    const int tid = threadIdx.x;
    float ax = 0.f, ay = 0.f;
    #pragma unroll
    for (int s = 0; s < 8; ++s) {
        const float2 p = *(const float2*)(partial + ((size_t)(b * 8 + s)) * ENC_ + tid * 2);
        ax += p.x; ay += p.y;
    }
    float2 o; o.x = ax; o.y = ay;
    *(float2*)(ctx + b * ENC_ + tid * 2) = o;
}

// ===========================================================================
extern "C" void kernel_launch(void* const* d_in, const int* in_sizes, int n_in,
                              void* d_out, int out_size, void* d_ws, size_t ws_size,
                              hipStream_t stream) {
    const float* dh  = (const float*)d_in[0];   // [B, DEC]
    const float* enc = (const float*)d_in[1];   // [B, T, ENC]
    const float* Wa  = (const float*)d_in[2];   // [DEC, U]
    const float* ba  = (const float*)d_in[3];   // [U]
    const float* Wb  = (const float*)d_in[4];   // [ENC, U]
    const float* bb  = (const float*)d_in[5];   // [U]
    const float* Wv  = (const float*)d_in[6];   // [U, 1]
    const float* bv  = (const float*)d_in[7];   // [1]

    char* ws = (char*)d_ws;
    float*          qq      = (float*)(ws + OFF_QQ);
    unsigned short* wbt     = (unsigned short*)(ws + OFF_WBT);
    float*          scoreb  = (float*)(ws + OFF_SCORE);
    float*          partial = (float*)(ws + OFF_PARTIAL);

    float* ctx  = (float*)d_out;                // [B, ENC] first
    float* attn = (float*)d_out + B_ * ENC_;    // [B, T, 1] second

    prep_q_kernel<<<B_, U_, 0, stream>>>(dh, Wa, ba, bb, qq);
    prep_wbt_kernel<<<(ENC_ * U_) / 256, 256, 0, stream>>>(Wb, wbt);

    score_wmma_kernel<<<dim3(T_ / TT, B_), 512, LDS_BYTES, stream>>>(
        enc, qq, Wv, bv, wbt, scoreb);

    softmax_kernel<<<B_, 256, 0, stream>>>(scoreb, attn);
    ctx_partial_kernel<<<dim3(8, B_), 256, 0, stream>>>(enc, attn, partial);
    ctx_reduce_kernel<<<B_, 256, 0, stream>>>(partial, ctx);
}
